// LSTM_31894427140527
// MI455X (gfx1250) — matrix operand order repaired and checked by reading, not compiled, not run on hardware
//
#include <hip/hip_runtime.h>
#include <math.h>

// Problem sizes (fixed by the reference)
#define B_SZ 32
#define T_SZ 512
#define D_SZ 1024
#define H_SZ 1024
#define G4H  4096          // 4*H
#define KT_D 32            // D / 32 (k-tiles for input GEMM)
#define KT_H 32            // H / 32 (k-tiles for recurrent GEMM)
#define NT_G 256           // 4H / 16 (n-tiles)
#define FRAG_ELEMS 512     // 32 lanes * 16 f16 per packed B fragment

typedef __attribute__((ext_vector_type(16))) _Float16 v16h;
typedef __attribute__((ext_vector_type(8)))  float    v8f;

union AFrag { v16h v; uint4 u[2]; };
union BFrag { v16h v; uint4 u[2]; };
union CFrag { v8f  v; float f[8]; };

// ---------------------------------------------------------------------------
// Kernel 1: convert X [B,T,D] f32 -> Xh [(t*B+b), D] f16 (time-major rows)
// ---------------------------------------------------------------------------
__global__ void k_convert_x(const float* __restrict__ X, _Float16* __restrict__ Xh) {
    size_t i = (size_t)blockIdx.x * blockDim.x + threadIdx.x;   // one per 4 elems
    size_t base = i * 4;
    size_t m = base / D_SZ;           // destination row: m = t*B + b
    int    d = (int)(base % D_SZ);
    int    t = (int)(m >> 5);         // m / 32
    int    b = (int)(m & 31);
    const float4 x = *(const float4*)(X + ((size_t)b * T_SZ + t) * D_SZ + d);
    union { _Float16 h[4]; uint2 u; } o;
    o.h[0] = (_Float16)x.x; o.h[1] = (_Float16)x.y;
    o.h[2] = (_Float16)x.z; o.h[3] = (_Float16)x.w;
    *(uint2*)(Xh + base) = o.u;
}

// ---------------------------------------------------------------------------
// Kernel 2: pack 4 concatenated [K=1024, H] f32 weight matrices into the exact
// wave32 WMMA B-fragment layout, f16.  Fragment f = ntile*KT + kt holds
// 32 lanes x 16 contiguous f16 (lane covers col N = ntile*16 + (lane&15),
// K = kt*32 + (e>>3)*16 + (lane>>4)*8 + (e&7)).
// ---------------------------------------------------------------------------
__global__ void k_pack_w(const float* __restrict__ W0, const float* __restrict__ W1,
                         const float* __restrict__ W2, const float* __restrict__ W3,
                         _Float16* __restrict__ Wp, int kTiles) {
    int wid  = blockIdx.x * (blockDim.x >> 5) + (threadIdx.x >> 5);  // fragment id
    int lane = threadIdx.x & 31;
    int ntile = wid / kTiles;
    int kt    = wid % kTiles;
    int col = lane & 15;
    int kb  = (lane >> 4) * 16;
    int N = ntile * 16 + col;
    const float* W = (N < H_SZ) ? W0 : (N < 2 * H_SZ) ? W1 : (N < 3 * H_SZ) ? W2 : W3;
    int n = N & (H_SZ - 1);
    union { _Float16 h[16]; uint4 u[2]; } frag;
#pragma unroll
    for (int e = 0; e < 16; ++e) {
        int K = kt * 32 + (e & 8) * 2 + (kb >> 1) + (e & 7);
        frag.h[e] = (_Float16)W[(size_t)K * H_SZ + n];
    }
    uint4* dst = (uint4*)(Wp + ((size_t)wid * 32 + lane) * 16);
    dst[0] = frag.u[0];
    dst[1] = frag.u[1];
}

// ---------------------------------------------------------------------------
// Kernel 3: zero cell state s and h ping buffer (workspace is poisoned 0xAA)
// ---------------------------------------------------------------------------
__global__ void k_init(float* __restrict__ s, _Float16* __restrict__ hhA) {
    int i = blockIdx.x * 256 + threadIdx.x;       // 32768 total
    s[i] = 0.0f;
    hhA[i] = (_Float16)0.0f;
}

// ---------------------------------------------------------------------------
// Kernel 4: xg[m, 0:4096] = Xh[m, :] @ Wxp + bias     (m = t*B + b, f32 out)
// One wave computes a 32x32 output tile: 2 m-tiles x 2 n-tiles, 4 WMMA
// accumulators; both A-frags and both B-frags are reused twice per k-step
// (16 flops per byte of L2 traffic vs 10.9 for a 16x32 tile).
// ---------------------------------------------------------------------------
__global__ void __launch_bounds__(256)
k_gemm_xg(const _Float16* __restrict__ Xh, const _Float16* __restrict__ Wxp,
          const float* __restrict__ ba, const float* __restrict__ bi,
          const float* __restrict__ bfv, const float* __restrict__ bo,
          float* __restrict__ xg) {
    const int wid  = blockIdx.x * 8 + (threadIdx.x >> 5);
    const int lane = threadIdx.x & 31;
    const int mp   = wid >> 7;            // 512 m-tile pairs (32 rows each)
    const int np   = wid & 127;           // 128 n-tile pairs (32 cols each)
    const int nt0  = np * 2;
    const int kofs = (lane >> 4) * 8;
    const _Float16* aRow0  = Xh + (size_t)(mp * 32 + (lane & 15)) * D_SZ;
    const _Float16* aRow1  = aRow0 + (size_t)16 * D_SZ;
    const _Float16* bBase0 = Wxp + (size_t)nt0 * KT_D * FRAG_ELEMS + (size_t)lane * 16;
    const _Float16* bBase1 = bBase0 + (size_t)KT_D * FRAG_ELEMS;

    CFrag c00, c01, c10, c11;
#pragma unroll
    for (int r = 0; r < 8; ++r) {
        c00.f[r] = 0.0f; c01.f[r] = 0.0f; c10.f[r] = 0.0f; c11.f[r] = 0.0f;
    }

    for (int kt = 0; kt < KT_D; ++kt) {
        AFrag a0, a1;
        const _Float16* ap0 = aRow0 + kt * 32 + kofs;
        const _Float16* ap1 = aRow1 + kt * 32 + kofs;
        a0.u[0] = *(const uint4*)(ap0);
        a0.u[1] = *(const uint4*)(ap0 + 16);
        a1.u[0] = *(const uint4*)(ap1);
        a1.u[1] = *(const uint4*)(ap1 + 16);
        BFrag b0, b1;
        const _Float16* bp0 = bBase0 + (size_t)kt * FRAG_ELEMS;
        const _Float16* bp1 = bBase1 + (size_t)kt * FRAG_ELEMS;
        b0.u[0] = *(const uint4*)(bp0);
        b0.u[1] = *(const uint4*)(bp0 + 8);
        b1.u[0] = *(const uint4*)(bp1);
        b1.u[1] = *(const uint4*)(bp1 + 8);
        __builtin_prefetch(bp0 + FRAG_ELEMS, 0, 1);   // gfx1250 global_prefetch_b8
        c00.v = __builtin_amdgcn_wmma_f32_16x16x32_f16(false, a0.v, false, b0.v,
                                                       (short)0, c00.v, false, false);
        c01.v = __builtin_amdgcn_wmma_f32_16x16x32_f16(false, a0.v, false, b1.v,
                                                       (short)0, c01.v, false, false);
        c10.v = __builtin_amdgcn_wmma_f32_16x16x32_f16(false, a1.v, false, b0.v,
                                                       (short)0, c10.v, false, false);
        c11.v = __builtin_amdgcn_wmma_f32_16x16x32_f16(false, a1.v, false, b1.v,
                                                       (short)0, c11.v, false, false);
    }

#pragma unroll
    for (int j = 0; j < 2; ++j) {
        int N = (nt0 + j) * 16 + (lane & 15);
        int gate = N >> 10;
        const float* bp = (gate == 0) ? ba : (gate == 1) ? bi : (gate == 2) ? bfv : bo;
        float bias = bp[N & (H_SZ - 1)];
#pragma unroll
        for (int i = 0; i < 2; ++i) {
            CFrag& c = (i == 0) ? ((j == 0) ? c00 : c01) : ((j == 0) ? c10 : c11);
            int mBase = mp * 32 + i * 16 + ((lane >> 4) << 3);
#pragma unroll
            for (int r = 0; r < 8; ++r)
                xg[(size_t)(mBase + r) * G4H + N] = c.f[r] + bias;
        }
    }
}

// ---------------------------------------------------------------------------
// Kernel 5: one LSTM timestep.
//   g = xg_t + h_{t-1} @ Wh   (WMMA, accumulator seeded from xg_t)
//   a=tanh, i/f/o=sigmoid, s = a*i + s*f, h = tanh(s)*o
// Grid: 32 blocks (32 hidden cols each) x 256 threads (8 waves).
// Wave w: gate = w>>1, n-subtile = w&1.  Each wave computes BOTH m-tiles of
// its 16-column slice (2 accumulators sharing one Wh B-fragment -> half the
// Wh L2 traffic per step).  Gate tiles meet in LDS for the cell update.
// h is double-buffered across steps (hh_in/hh_out).
// ---------------------------------------------------------------------------
__global__ void __launch_bounds__(256)
k_step(const float* __restrict__ xg_t, const _Float16* __restrict__ Whp,
       const _Float16* __restrict__ hh_in, _Float16* __restrict__ hh_out,
       float* __restrict__ s, float* __restrict__ out, int t) {
    __shared__ float sm[4][32][32];              // [gate][batch][local col] 16 KB

    const int wave = threadIdx.x >> 5;
    const int lane = threadIdx.x & 31;
    const int gate = wave >> 1;
    const int nsub = wave & 1;
    const int jb   = blockIdx.x;                 // 0..31: hidden cols jb*32..+31
    const int nt   = gate * 64 + jb * 2 + nsub;  // global n-tile in [0,256)
    const int col  = lane & 15;
    const int N    = nt * 16 + col;
    const int kofs = (lane >> 4) * 8;
    const int rBase = (lane >> 4) << 3;

    CFrag c0, c1;                                // m-tiles 0 (rows 0-15), 1 (16-31)
#pragma unroll
    for (int r = 0; r < 8; ++r) {
        c0.f[r] = xg_t[(size_t)(rBase + r) * G4H + N];
        c1.f[r] = xg_t[(size_t)(16 + rBase + r) * G4H + N];
    }

    const _Float16* aRow0 = hh_in + (size_t)col * H_SZ;
    const _Float16* aRow1 = aRow0 + (size_t)16 * H_SZ;
    const _Float16* bBase = Whp + (size_t)nt * KT_H * FRAG_ELEMS + (size_t)lane * 16;

    for (int kt = 0; kt < KT_H; ++kt) {
        AFrag a0, a1;
        const _Float16* ap0 = aRow0 + kt * 32 + kofs;
        const _Float16* ap1 = aRow1 + kt * 32 + kofs;
        a0.u[0] = *(const uint4*)(ap0);
        a0.u[1] = *(const uint4*)(ap0 + 16);
        a1.u[0] = *(const uint4*)(ap1);
        a1.u[1] = *(const uint4*)(ap1 + 16);
        BFrag b;
        const _Float16* bp = bBase + (size_t)kt * FRAG_ELEMS;
        b.u[0] = *(const uint4*)(bp);
        b.u[1] = *(const uint4*)(bp + 8);
        __builtin_prefetch(bp + FRAG_ELEMS, 0, 1);
        c0.v = __builtin_amdgcn_wmma_f32_16x16x32_f16(false, a0.v, false, b.v,
                                                      (short)0, c0.v, false, false);
        c1.v = __builtin_amdgcn_wmma_f32_16x16x32_f16(false, a1.v, false, b.v,
                                                      (short)0, c1.v, false, false);
    }

    const int lcol = nsub * 16 + col;            // local col within block window
#pragma unroll
    for (int r = 0; r < 8; ++r) {
        sm[gate][rBase + r][lcol]      = c0.f[r];
        sm[gate][16 + rBase + r][lcol] = c1.f[r];
    }
    __syncthreads();

    for (int idx = threadIdx.x; idx < 1024; idx += 256) {
        int m = idx >> 5;                        // batch row 0..31
        int j = idx & 31;                        // local hidden col
        float av = tanhf(sm[0][m][j]);
        float iv = 1.0f / (1.0f + __expf(-sm[1][m][j]));
        float fv = 1.0f / (1.0f + __expf(-sm[2][m][j]));
        float ov = 1.0f / (1.0f + __expf(-sm[3][m][j]));
        int hcol = jb * 32 + j;
        size_t sidx = (size_t)m * H_SZ + hcol;
        float sNew = av * iv + s[sidx] * fv;
        float hv = tanhf(sNew) * ov;
        s[sidx] = sNew;
        out[((size_t)m * T_SZ + t) * H_SZ + hcol] = hv;
        hh_out[sidx] = (_Float16)hv;
    }
}

// ---------------------------------------------------------------------------
extern "C" void kernel_launch(void* const* d_in, const int* in_sizes, int n_in,
                              void* d_out, int out_size, void* d_ws, size_t ws_size,
                              hipStream_t stream) {
    const float* X    = (const float*)d_in[0];
    const float* Wa_x = (const float*)d_in[1];
    const float* Wi_x = (const float*)d_in[2];
    const float* Wf_x = (const float*)d_in[3];
    const float* Wo_x = (const float*)d_in[4];
    const float* Wa_h = (const float*)d_in[5];
    const float* Wi_h = (const float*)d_in[6];
    const float* Wf_h = (const float*)d_in[7];
    const float* Wo_h = (const float*)d_in[8];
    const float* ba   = (const float*)d_in[9];
    const float* bi   = (const float*)d_in[10];
    const float* bfv  = (const float*)d_in[11];
    const float* bo   = (const float*)d_in[12];
    float* out = (float*)d_out;

    // Carve workspace (256B-aligned slabs)
    char* ws = (char*)d_ws;
    auto carve = [&](size_t bytes) {
        char* p = ws;
        ws += (bytes + 255) & ~(size_t)255;
        return p;
    };
    _Float16* Xh  = (_Float16*)carve((size_t)T_SZ * B_SZ * D_SZ * 2);   // 32 MB
    _Float16* Wxp = (_Float16*)carve((size_t)D_SZ * G4H * 2);           //  8 MB
    _Float16* Whp = (_Float16*)carve((size_t)H_SZ * G4H * 2);           //  8 MB
    float*    xg  = (float*)   carve((size_t)T_SZ * B_SZ * G4H * 4);    // 256 MB
    _Float16* hhA = (_Float16*)carve((size_t)B_SZ * H_SZ * 2);
    _Float16* hhB = (_Float16*)carve((size_t)B_SZ * H_SZ * 2);
    float*    s   = (float*)   carve((size_t)B_SZ * H_SZ * 4);

    // Phase 0: conversions / weight packing / state init
    k_convert_x<<<(B_SZ * T_SZ * D_SZ / 4) / 256, 256, 0, stream>>>(X, Xh);
    k_pack_w<<<(NT_G * KT_D) / 8, 256, 0, stream>>>(Wa_x, Wi_x, Wf_x, Wo_x, Wxp, KT_D);
    k_pack_w<<<(NT_G * KT_H) / 8, 256, 0, stream>>>(Wa_h, Wi_h, Wf_h, Wo_h, Whp, KT_H);
    k_init<<<(B_SZ * H_SZ) / 256, 256, 0, stream>>>(s, hhA);

    // Phase 1: fused input projection for all timesteps (time-major xg)
    // waves = 512 m-pairs * 128 n-pairs = 65536 -> 8192 blocks of 8 waves
    k_gemm_xg<<<8192, 256, 0, stream>>>(Xh, Wxp, ba, bi, bfv, bo, xg);

    // Phase 2: sequential recurrence, one launch per timestep (stream-ordered)
    for (int t = 0; t < T_SZ; ++t) {
        const _Float16* hin = (t & 1) ? hhB : hhA;
        _Float16*       hout = (t & 1) ? hhA : hhB;
        k_step<<<32, 256, 0, stream>>>(xg + (size_t)t * B_SZ * G4H, Whp,
                                       hin, hout, s, out, t);
    }
}
